// Decoder_19129784336793
// MI455X (gfx1250) — matrix operand order repaired and checked by reading, not compiled, not run on hardware
//
#include <hip/hip_runtime.h>

// ---------------------------------------------------------------------------
// PWC-Net decoder level for MI455X (gfx1250, wave32, WMMA).
// Conv stack = implicit GEMM on v_wmma_f32_16x16x32_f16 (f32 accumulate).
// Dense concats are free via a channel-offset feat layout built in d_out.
// ---------------------------------------------------------------------------

typedef __attribute__((ext_vector_type(16))) _Float16 v16h;
typedef __attribute__((ext_vector_type(8)))  float    v8f;

namespace {
constexpr int   Bn    = 2;
constexpr int   H     = 192, W = 192, HW = H * W;
constexpr int   hS    = 96,  wS = 96, hwS = hS * wS;
constexpr int   FEATC = 550;
// final feat channel layout: [o4(16) o3(32) o2(64) o1(96) o0(128) base(214)]
constexpr int   BASE_VOL    = 336;   // 81 ch
constexpr int   BASE_FIRST  = 417;   // 128 ch
constexpr int   BASE_FLOW   = 545;   // 2 ch
constexpr int   BASE_UPFEAT = 547;   // 2 ch
constexpr int   BASE_MASK   = 549;   // 1 ch
constexpr int   MAXCINPAD   = 576;   // 18 chunks * 32 (conv5: Cin=550)
constexpr float DBL_BACKWARP = 0.625f;
}

__device__ __forceinline__ float leaky(float v) { return v > 0.f ? v : 0.1f * v; }

// --------------------------- 2x bilinear upsample (align_corners=True) -----
__global__ void upsample2x_kernel(const float* __restrict__ in, float* __restrict__ out,
                                  int C, long out_bstride, int out_coff)
{
    long i = (long)blockIdx.x * blockDim.x + threadIdx.x;
    long total = (long)Bn * C * HW;
    if (i >= total) return;
    int x = (int)(i % W);
    int y = (int)((i / W) % H);
    int c = (int)((i / HW) % C);
    int b = (int)(i / ((long)C * HW));
    float py = y * (float)(hS - 1) / (float)(H - 1);
    float px = x * (float)(wS - 1) / (float)(W - 1);
    int y0 = (int)py, x0 = (int)px;
    int y1 = min(y0 + 1, hS - 1), x1 = min(x0 + 1, wS - 1);
    float fy = py - y0, fx = px - x0;
    const float* p = in + ((long)b * C + c) * hwS;
    float v = (1.f - fy) * ((1.f - fx) * p[y0 * wS + x0] + fx * p[y0 * wS + x1])
            +        fy  * ((1.f - fx) * p[y1 * wS + x0] + fx * p[y1 * wS + x1]);
    out[(long)b * out_bstride + (long)(out_coff + c) * HW + (long)y * W + x] = v;
}

// --------------------------- ConvTranspose2d(k=4,s=2,p=1), Cout=2 ----------
__global__ void deconv4s2_kernel(const float* __restrict__ in, const float* __restrict__ wt,
                                 const float* __restrict__ bias, int Cin,
                                 float* __restrict__ out, long out_bstride, int out_coff)
{
    long i = (long)blockIdx.x * blockDim.x + threadIdx.x;
    long total = (long)Bn * 2 * HW;
    if (i >= total) return;
    int x  = (int)(i % W);
    int y  = (int)((i / W) % H);
    int oc = (int)((i / HW) % 2);
    int b  = (int)(i / (2L * HW));
    float acc = bias[oc];
    const float* ib = in + (long)b * Cin * hwS;
    for (int ky = 0; ky < 4; ++ky) {
        int t = y + 1 - ky;
        if (t < 0 || (t & 1)) continue;
        int iy = t >> 1;
        if (iy >= hS) continue;
        for (int kx = 0; kx < 4; ++kx) {
            int s = x + 1 - kx;
            if (s < 0 || (s & 1)) continue;
            int ix = s >> 1;
            if (ix >= wS) continue;
            const float* ip = ib + (long)iy * wS + ix;
            const float* wp = wt + (ky * 4 + kx) + oc * 16;   // w[ic][oc][ky][kx]
            for (int ic = 0; ic < Cin; ++ic)
                acc += wp[(long)ic * 32] * ip[(long)ic * hwS];
        }
    }
    out[(long)b * out_bstride + (long)(out_coff + oc) * HW + (long)y * W + x] = acc;
}

// --------------------------- mask passthrough ------------------------------
__global__ void copy_mask_kernel(const float* __restrict__ mask, float* __restrict__ featbase)
{
    long i = (long)blockIdx.x * blockDim.x + threadIdx.x;
    if (i >= (long)Bn * HW) return;
    long b = i / HW, r = i % HW;
    featbase[b * (long)FEATC * HW + (long)BASE_MASK * HW + r] = mask[i];
}

// --------------------------- flow warp (grid_sample zeros + validity mask) -
__global__ void warp_kernel(const float* __restrict__ second, const float* __restrict__ featbase,
                            float* __restrict__ warped)
{
    long i = (long)blockIdx.x * blockDim.x + threadIdx.x;
    long total = (long)Bn * 128 * HW;
    if (i >= total) return;
    int x = (int)(i % W);
    int y = (int)((i / W) % H);
    int c = (int)((i / HW) % 128);
    int b = (int)(i / (128L * HW));
    const float* fb = featbase + (long)b * FEATC * HW;
    float fx = fb[(long)BASE_FLOW * HW + (long)y * W + x] * DBL_BACKWARP;
    float fy = fb[(long)(BASE_FLOW + 1) * HW + (long)y * W + x] * DBL_BACKWARP;
    float gx = 2.f * (x + fx) / (float)(W - 1) - 1.f;
    float gy = 2.f * (y + fy) / (float)(H - 1) - 1.f;
    float ix = ((gx + 1.f) * W - 1.f) * 0.5f;
    float iy = ((gy + 1.f) * H - 1.f) * 0.5f;
    int ix0 = (int)floorf(ix), iy0 = (int)floorf(iy);
    float wx1 = ix - ix0, wy1 = iy - iy0;
    float wx0 = 1.f - wx1, wy0 = 1.f - wy1;
    const float* p = second + ((long)b * 128 + c) * HW;
    float o = 0.f, m = 0.f;
    int ys[2] = { iy0, iy0 + 1 };
    int xs[2] = { ix0, ix0 + 1 };
    float wy[2] = { wy0, wy1 };
    float wx[2] = { wx0, wx1 };
    #pragma unroll
    for (int a = 0; a < 2; ++a)
        #pragma unroll
        for (int d = 0; d < 2; ++d) {
            int yy = ys[a], xx = xs[d];
            if (yy >= 0 && yy < H && xx >= 0 && xx < W) {
                float wgt = wy[a] * wx[d];
                o += wgt * p[(long)yy * W + xx];
                m += wgt;
            }
        }
    float msk = (m >= 0.9999f) ? 1.f : 0.f;
    warped[((long)b * 128 + c) * HW + (long)y * W + x] = o * msk;
}

// --------------------------- 81-tap local correlation + leaky --------------
__global__ void corr_kernel(float* __restrict__ featbase, const float* __restrict__ warped)
{
    long i = (long)blockIdx.x * blockDim.x + threadIdx.x;
    if (i >= (long)Bn * HW) return;
    int x = (int)(i % W);
    int y = (int)((i / W) % H);
    int b = (int)(i / HW);
    const float* ref = featbase + (long)b * FEATC * HW + (long)BASE_FIRST * HW + (long)y * W + x;
    const float* tgt = warped + (long)b * 128 * HW;
    float* vol = featbase + (long)b * FEATC * HW + (long)BASE_VOL * HW + (long)y * W + x;
    const float inv = 1.f / 128.f;
    for (int di = 0; di < 9; ++di) {
        float acc[9];
        #pragma unroll
        for (int dj = 0; dj < 9; ++dj) acc[dj] = 0.f;
        int yy = y + di - 4;
        if (yy >= 0 && yy < H) {
            for (int c = 0; c < 128; ++c) {
                float rv = ref[(long)c * HW];
                const float* trow = tgt + (long)c * HW + (long)yy * W;
                #pragma unroll
                for (int dj = 0; dj < 9; ++dj) {
                    int xx = x + dj - 4;
                    if (xx >= 0 && xx < W) acc[dj] += rv * trow[xx];
                }
            }
        }
        #pragma unroll
        for (int dj = 0; dj < 9; ++dj)
            vol[(long)(di * 9 + dj) * HW] = leaky(acc[dj] * inv);
    }
}

// --------------------------- weight pre-pack: fp32 OIHW -> f16 A-fragments -
// Packed layout: frag = ((g*9 + tap)*nchunk + chunk); element (lane,e)
// uses contiguous k order K = 16*(lane/16) + e, M = lane%16 (as the B frag), so each lane's fragment is a
// contiguous 32B load.
__global__ void pack_weights_kernel(const float* __restrict__ wt, _Float16* __restrict__ o,
                                    int cin, int cout, int nchunk, int ngrp)
{
    long i = (long)blockIdx.x * blockDim.x + threadIdx.x;
    long total = (long)ngrp * 9 * nchunk * 512;
    if (i >= total) return;
    int  e    = (int)(i & 15);
    int  lane = (int)((i >> 4) & 31);
    long q    = i >> 9;
    int  chunk = (int)(q % nchunk); q /= nchunk;
    int  tap   = (int)(q % 9);
    int  g     = (int)(q / 9);
    int M = lane & 15;
    int K = (lane < 16) ? e                           // K = {0..15}
                        : e + 16;                     // K = {16..31}
    int oc = g * 16 + M;
    int ic = chunk * 32 + K;
    float v = 0.f;
    if (oc < cout && ic < cin)
        v = wt[((long)oc * cin + ic) * 9 + tap];      // OIHW, tap = ky*3+kx
    o[i] = (_Float16)v;
}

// --------------------------- implicit-GEMM 3x3 conv on WMMA ----------------
// Block = 256 threads (8 waves), 16-pixel row tile. LDS holds the f16
// activation halo tile [3 rows][18 cols][cinPad ch] so each lane's B fragment
// (16 consecutive channels) is one contiguous 32B ds read. Wave w computes
// output channels [16w, 16w+16) for the same 16 pixels.
__global__ __launch_bounds__(256) void conv3x3_wmma_kernel(
    const float* __restrict__ featbase, int cin_off, int cin, int nchunk,
    const _Float16* __restrict__ wpk, const float* __restrict__ bias,
    int ngrp, int cout, float* __restrict__ outp, long out_bstride, int out_coff)
{
    __shared__ __attribute__((aligned(32))) _Float16 lds[MAXCINPAD * 54];
    const int cinPad = nchunk * 32;
    const int x0 = blockIdx.x * 16;
    const int y  = blockIdx.y;
    const int b  = blockIdx.z;
    const int tid = threadIdx.x;

    const float* fb = featbase + (long)b * FEATC * HW + (long)cin_off * HW;
    for (int i = tid; i < cinPad * 54; i += 256) {
        int c = i / 54, r = i % 54;          // r = ry*18 + cx
        int ry = r / 18, cx = r % 18;
        int yy = y + ry - 1, xx = x0 + cx - 1;
        float v = 0.f;
        if (c < cin && yy >= 0 && yy < H && xx >= 0 && xx < W)
            v = fb[(long)c * HW + (long)yy * W + xx];
        lds[(long)r * cinPad + c] = (_Float16)v;
    }
    __syncthreads();

    const int wid  = tid >> 5;
    const int lane = tid & 31;
    if (wid >= ngrp) return;

    v8f acc = {};
    const int n   = lane & 15;          // pixel within tile (N)
    const int chi = (lane >> 4) << 4;   // B-frag channel half: 0 or 16
    for (int tap = 0; tap < 9; ++tap) {
        const int ky = tap / 3, kx = tap % 3;
        const _Float16* arow = wpk + ((long)(wid * 9 + tap) * nchunk) * 512 + (long)lane * 16;
        const _Float16* brow = lds + (long)(ky * 18 + n + kx) * cinPad + chi;
        for (int chunk = 0; chunk < nchunk; ++chunk) {
            v16h a  = *(const v16h*)(arow + (long)chunk * 512);
            v16h bf = *(const v16h*)(brow + chunk * 32);
            acc = __builtin_amdgcn_wmma_f32_16x16x32_f16(
                false, a, false, bf, (short)0, acc, false, false);
        }
    }

    // D layout: VGPR r -> M = r + 8*(lane>=16); N = lane&15
    const int mhi = (lane >> 4) * 8;
    float* op = outp + (long)b * out_bstride + (long)out_coff * HW + (long)y * W + x0 + n;
    #pragma unroll
    for (int r = 0; r < 8; ++r) {
        int oc = wid * 16 + mhi + r;
        if (oc < cout)
            op[(long)oc * HW] = leaky(acc[r] + bias[oc]);
    }
}

// ---------------------------------------------------------------------------
extern "C" void kernel_launch(void* const* d_in, const int* in_sizes, int n_in,
                              void* d_out, int out_size, void* d_ws, size_t ws_size,
                              hipStream_t stream)
{
    (void)in_sizes; (void)n_in; (void)out_size; (void)ws_size;
    const float* tensorFirst  = (const float*)d_in[0];
    const float* tensorSecond = (const float*)d_in[1];
    const float* prev_flow    = (const float*)d_in[2];
    const float* prev_feat    = (const float*)d_in[3];
    const float* mask         = (const float*)d_in[4];
    const float* w_upflow     = (const float*)d_in[5];
    const float* b_upflow     = (const float*)d_in[6];
    const float* w_upfeat     = (const float*)d_in[7];
    const float* b_upfeat     = (const float*)d_in[8];
    const float* wconv[6]; const float* bconv[6];
    for (int i = 0; i < 6; ++i) { wconv[i] = (const float*)d_in[9 + 2 * i];
                                  bconv[i] = (const float*)d_in[10 + 2 * i]; }

    float* flow_out = (float*)d_out;                         // [B,2,192,192]
    float* feat     = flow_out + (size_t)Bn * 2 * HW;        // [B,550,192,192]

    float*     second_up = (float*)d_ws;                     // [B,128,192,192]
    float*     warped    = second_up + (size_t)Bn * 128 * HW;
    _Float16*  wpk_base  = (_Float16*)(warped + (size_t)Bn * 128 * HW);

    const int T = 256;
    // Upsampled first goes straight into the feat buffer (free concat).
    {
        long total = (long)Bn * 128 * HW;
        int blk = (int)((total + T - 1) / T);
        upsample2x_kernel<<<blk, T, 0, stream>>>(tensorFirst, feat, 128, (long)FEATC * HW, BASE_FIRST);
        upsample2x_kernel<<<blk, T, 0, stream>>>(tensorSecond, second_up, 128, (long)128 * HW, 0);
    }
    {
        long total = (long)Bn * 2 * HW;
        int blk = (int)((total + T - 1) / T);
        deconv4s2_kernel<<<blk, T, 0, stream>>>(prev_flow, w_upflow, b_upflow, 2,
                                                feat, (long)FEATC * HW, BASE_FLOW);
        deconv4s2_kernel<<<blk, T, 0, stream>>>(prev_feat, w_upfeat, b_upfeat, 418,
                                                feat, (long)FEATC * HW, BASE_UPFEAT);
    }
    {
        long total = (long)Bn * HW;
        int blk = (int)((total + T - 1) / T);
        copy_mask_kernel<<<blk, T, 0, stream>>>(mask, feat);
    }
    {
        long total = (long)Bn * 128 * HW;
        int blk = (int)((total + T - 1) / T);
        warp_kernel<<<blk, T, 0, stream>>>(second_up, feat, warped);
    }
    {
        long total = (long)Bn * HW;
        int blk = (int)((total + T - 1) / T);
        corr_kernel<<<blk, T, 0, stream>>>(feat, warped);
    }

    // Dense conv stack: channel-suffix reads, prefix writes.
    const int cins  [6] = { 214, 342, 438, 502, 534, 550 };
    const int couts [6] = { 128,  96,  64,  32,  16,   2 };
    const int cinoff[6] = { 336, 208, 112,  48,  16,   0 };
    const int outoff[6] = { 208, 112,  48,  16,   0,   0 };
    int nchunks[6], ngrps[6];
    _Float16* wp[6];
    size_t woff = 0;
    for (int i = 0; i < 6; ++i) {
        nchunks[i] = (cins[i] + 31) / 32;
        ngrps[i]   = (couts[i] + 15) / 16;
        wp[i] = wpk_base + woff;
        long total = (long)ngrps[i] * 9 * nchunks[i] * 512;
        pack_weights_kernel<<<(int)((total + T - 1) / T), T, 0, stream>>>(
            wconv[i], wp[i], cins[i], couts[i], nchunks[i], ngrps[i]);
        woff += (size_t)total;
    }

    dim3 cgrid(W / 16, H, Bn);
    for (int i = 0; i < 6; ++i) {
        float* op; long bstride; int coff;
        if (i < 5) { op = feat;     bstride = (long)FEATC * HW; coff = outoff[i]; }
        else       { op = flow_out; bstride = (long)2 * HW;     coff = 0; }
        conv3x3_wmma_kernel<<<cgrid, 256, 0, stream>>>(
            feat, cinoff[i], cins[i], nchunks[i],
            wp[i], bconv[i], ngrps[i], couts[i], op, bstride, coff);
    }
}
